// MedicalCodingModel_13116830122169
// MI455X (gfx1250) — matrix operand order repaired and checked by reading, not compiled, not run on hardware
//
#include <hip/hip_runtime.h>
#include <hip/hip_bf16.h>
#include <math.h>

#define S_LEN 4096
#define HID 768
#define NHEAD 12
#define DHEAD 64
#define NLAYER 2
#define WWIN 256
#define NCHUNK 16
#define FF 3072
#define NUM_ICD 8000
#define NUM_CPT 5000
#define NUM_CH 22

typedef __bf16 bf16_t;
typedef __attribute__((ext_vector_type(16))) __bf16 v16bf;
typedef __attribute__((ext_vector_type(8))) float v8f;

// ---------------------------------------------------------------------------
// WMMA helpers (CDNA5 gfx1250, wave32): D(16x16 f32) = A(16x32 bf16) x B(32x16 bf16) + C
// ---------------------------------------------------------------------------
__device__ inline v8f wmma_bf16(v16bf a, v16bf b, v8f c) {
  return __builtin_amdgcn_wmma_f32_16x16x32_bf16(false, a, false, b, (short)0, c,
                                                 false, false);
}

// A-fragment from LDS: tile stored row-major [16 rows][32 k], stride in elems.
// lanes 0-15 rows 0-15 hold K=0..7 / K=16..23; lanes 16-31 K=8..15 / K=24..31.
__device__ inline v16bf ldsA_frag(const bf16_t* base, int stride) {
  const int lane = threadIdx.x & 31;
  const int r = lane & 15;
  const int kb = (lane >> 4) * 8;
  const bf16_t* p = base + r * stride + kb;
  v16bf a;
#pragma unroll
  for (int e = 0; e < 8; ++e) { a[e] = p[e]; a[e + 8] = p[16 + e]; }
  return a;
}

// B-fragment from LDS: tile stored transposed [16 n][32 k], stride in elems.
// lanes 0-15 cols 0-15 hold K=0..7 / K=16..23; lanes 16-31 K=8..15 / K=24..31.
__device__ inline v16bf ldsB_frag(const bf16_t* base, int stride) {
  const int lane = threadIdx.x & 31;
  const int n = lane & 15;
  const int kb = (lane >> 4) * 8;
  const bf16_t* p = base + n * stride + kb;
  v16bf b;
#pragma unroll
  for (int e = 0; e < 8; ++e) { b[e] = p[e]; b[e + 8] = p[16 + e]; }
  return b;
}

// tanh-GELU via v_exp_f32 only: tanh(z) = 1 - 2/(exp(2z)+1); saturates cleanly.
__device__ inline float gelu_f(float x) {
  const float z2 = 1.5957691216057308f * (x + 0.044715f * x * x * x);  // 2*0.7978845608*(...)
  const float e = __expf(z2);
  const float t = 1.f - 2.f / (e + 1.f);
  return 0.5f * x * (1.f + t);
}

__device__ inline float block_sum(float v, float* red) {
  const int tid = threadIdx.x;
  red[tid] = v;
  __syncthreads();
#pragma unroll
  for (int off = 128; off > 0; off >>= 1) {
    if (tid < off) red[tid] += red[tid + off];
    __syncthreads();
  }
  float r = red[0];
  __syncthreads();
  return r;
}

// ---------------------------------------------------------------------------
// Generic GEMM: Y[M,N] = act(X[M,K] @ W[K,N] + bias). f32 in/out, bf16 WMMA.
// Block tile 128x128, K-step 32, 256 threads = 8 waves (4x2, 32x64 each).
// Register-prefetch + double-buffered LDS pipeline; coalesced global reads.
// ---------------------------------------------------------------------------
__global__ __launch_bounds__(256) void gemm_kernel(
    const float* __restrict__ X, const float* __restrict__ Wt,
    const float* __restrict__ bias, float* __restrict__ Y,
    int M, int N, int K, int act) {
  __shared__ bf16_t As[2][128 * 40];  // [row][k], padded stride 40
  __shared__ bf16_t Bs[2][128 * 40];  // [n][k] (B transposed), padded stride 40
  const int m0 = blockIdx.y * 128;
  const int n0 = blockIdx.x * 128;
  const int tid = threadIdx.x;
  const int wave = tid >> 5;
  const int lane = tid & 31;
  const int wm = wave & 3;   // 4 row groups of 32
  const int wn = wave >> 2;  // 2 col groups of 64

  // A staging: thread -> row (0..127), k-half {0,16}: contiguous 16-float reads.
  const int arow = tid >> 1;
  const int akh = (tid & 1) * 16;
  // B staging: thread -> even k pair (0..30), n group of 8: contiguous 8-float
  // reads from two adjacent W rows; packed bf16-pair LDS stores.
  const int bkp = (tid >> 4) * 2;
  const int bng = (tid & 15) * 8;

  float Ar[16], Br0[8], Br1[8];

  auto load_tile = [&](int k0) {
    const float* ga = X + (size_t)(m0 + arow) * K + k0 + akh;
#pragma unroll
    for (int e = 0; e < 16; ++e) Ar[e] = ga[e];
    const float* gb0 = Wt + (size_t)(k0 + bkp) * N + n0 + bng;
    const float* gb1 = gb0 + N;
#pragma unroll
    for (int e = 0; e < 8; ++e) { Br0[e] = gb0[e]; Br1[e] = gb1[e]; }
  };
  auto store_tile = [&](int p) {
    bf16_t* da = &As[p][arow * 40 + akh];
#pragma unroll
    for (int e = 0; e < 16; ++e) da[e] = (bf16_t)Ar[e];
#pragma unroll
    for (int e = 0; e < 8; ++e) {
      bf16_t* db = &Bs[p][(bng + e) * 40 + bkp];
      db[0] = (bf16_t)Br0[e];
      db[1] = (bf16_t)Br1[e];
    }
  };

  const v8f vzero = {};
  v8f acc[2][4];
#pragma unroll
  for (int i = 0; i < 2; ++i)
#pragma unroll
    for (int j = 0; j < 4; ++j) acc[i][j] = vzero;

  const int NT = K >> 5;
  load_tile(0);
  store_tile(0);
  __syncthreads();
  int p = 0;
#pragma unroll 1
  for (int kt = 0; kt < NT; ++kt) {
    if (kt + 1 < NT) load_tile((kt + 1) << 5);  // global loads in flight...
    v16bf af[2], bfr[4];
#pragma unroll
    for (int i = 0; i < 2; ++i)
      af[i] = ldsA_frag(&As[p][(wm * 32 + i * 16) * 40], 40);
#pragma unroll
    for (int j = 0; j < 4; ++j)
      bfr[j] = ldsB_frag(&Bs[p][(wn * 64 + j * 16) * 40], 40);
#pragma unroll
    for (int i = 0; i < 2; ++i)
#pragma unroll
      for (int j = 0; j < 4; ++j) acc[i][j] = wmma_bf16(af[i], bfr[j], acc[i][j]);
    if (kt + 1 < NT) store_tile(p ^ 1);         // ...while WMMAs issue
    __syncthreads();
    p ^= 1;
  }

  // epilogue: C layout -> lane holds col nl, rows (hi*8 + r)
  const int nl = lane & 15;
  const int hi = lane >> 4;
#pragma unroll
  for (int i = 0; i < 2; ++i)
#pragma unroll
    for (int j = 0; j < 4; ++j) {
      const int col = n0 + wn * 64 + j * 16 + nl;
      const float bv = bias[col];
#pragma unroll
      for (int r = 0; r < 8; ++r) {
        const int row = m0 + wm * 32 + i * 16 + hi * 8 + r;
        float vv = acc[i][j][r] + bv;
        if (act) vv = gelu_f(vv);
        Y[(size_t)row * N + col] = vv;
      }
    }
}

// ---------------------------------------------------------------------------
// Fused banded flash attention. grid = (2, NCHUNK, NHEAD), 256 threads = 8 waves.
// Block handles 128 query rows of chunk `nch`, head `h`. Band = 3W = 768 keys,
// staged in 6 slabs of 128 rows. Each wave owns one 16-row query tile.
// ---------------------------------------------------------------------------
__global__ __launch_bounds__(256) void band_attn_kernel(
    const float* __restrict__ qf, const float* __restrict__ kf,
    const float* __restrict__ vf, const float* __restrict__ amask,
    float* __restrict__ ctx) {
  __shared__ bf16_t Ks[128 * 72];     // [band row][d], stride 72
  __shared__ bf16_t Vs[64 * 136];     // [d][band row] (transposed), stride 136
  __shared__ bf16_t Ps[8 * 16 * 40];  // per-wave P tile 16x32, stride 40
  __shared__ float Mk[3 * WWIN];      // per-band-col sequence/attention mask

  const int sub = blockIdx.x;  // 0..1 (query half-chunk)
  const int nch = blockIdx.y;  // 0..15
  const int h = blockIdx.z;    // 0..11
  const int tid = threadIdx.x;
  const int w = tid >> 5;
  const int lane = tid & 31;
  const int nl = lane & 15;
  const int hi = lane >> 4;
  const float scale = 0.125f;  // 1/sqrt(64)

  // band-column mask (sequence range + attention_mask), computed once
  for (int i = tid; i < 3 * WWIN; i += 256) {
    const int kpos = nch * WWIN - WWIN + i;
    Mk[i] = (kpos >= 0 && kpos < S_LEN) ? amask[kpos] : 0.f;
  }

  // Q fragments (A layout), d split into two 32-wide WMMA K-steps
  v16bf qfr[2];
  {
    const int qrow = nch * WWIN + sub * 128 + w * 16 + nl;
    const float* qg = qf + (size_t)qrow * HID + h * DHEAD;
    const int kb8 = hi * 8;
#pragma unroll
    for (int st = 0; st < 2; ++st)
#pragma unroll
      for (int e = 0; e < 8; ++e) {
        qfr[st][e] = (bf16_t)qg[st * 32 + kb8 + e];
        qfr[st][e + 8] = (bf16_t)qg[st * 32 + kb8 + 16 + e];
      }
  }

  const v8f vzero = {};
  v8f o[4];
#pragma unroll
  for (int j = 0; j < 4; ++j) o[j] = vzero;
  float rm[8], rs[8];
#pragma unroll
  for (int r = 0; r < 8; ++r) { rm[r] = -1.0e30f; rs[r] = 0.f; }

  for (int c = 0; c < 6; ++c) {
    __syncthreads();
    {  // stage K slab + V slab (transposed, packed row-pairs), zero-fill OOB
      const int rp2 = (tid >> 2) * 2;   // even band row
      const int dg = (tid & 3) * 16;    // d group of 16
      const int kb = nch * WWIN - WWIN + c * 128;
      const int kpos0 = kb + rp2, kpos1 = kb + rp2 + 1;
      const bool in0 = (kpos0 >= 0) && (kpos0 < S_LEN);
      const bool in1 = (kpos1 >= 0) && (kpos1 < S_LEN);
      const float* kg0 = kf + (size_t)(in0 ? kpos0 : 0) * HID + h * DHEAD + dg;
      const float* kg1 = kf + (size_t)(in1 ? kpos1 : 0) * HID + h * DHEAD + dg;
      const float* vg0 = vf + (size_t)(in0 ? kpos0 : 0) * HID + h * DHEAD + dg;
      const float* vg1 = vf + (size_t)(in1 ? kpos1 : 0) * HID + h * DHEAD + dg;
#pragma unroll
      for (int e = 0; e < 16; ++e) {
        Ks[rp2 * 72 + dg + e] = (bf16_t)(in0 ? kg0[e] : 0.f);
        Ks[(rp2 + 1) * 72 + dg + e] = (bf16_t)(in1 ? kg1[e] : 0.f);
        bf16_t* dv = &Vs[(dg + e) * 136 + rp2];
        dv[0] = (bf16_t)(in0 ? vg0[e] : 0.f);
        dv[1] = (bf16_t)(in1 ? vg1[e] : 0.f);
      }
    }
    __syncthreads();

#pragma unroll 1
    for (int kc = 0; kc < 4; ++kc) {
      // ---- scores for 32 band cols (two 16x16 tiles) via WMMA over d
      v8f s[2];
#pragma unroll
      for (int st = 0; st < 2; ++st) {
        v16bf b0 = ldsB_frag(&Ks[(kc * 32 + st * 16) * 72 + 0], 72);
        v16bf b1 = ldsB_frag(&Ks[(kc * 32 + st * 16) * 72 + 32], 72);
        v8f a = vzero;
        a = wmma_bf16(qfr[0], b0, a);
        a = wmma_bf16(qfr[1], b1, a);
        s[st] = a;
      }
      // ---- band + sequence masking, scale (col mask hoisted out of r loop)
#pragma unroll
      for (int st = 0; st < 2; ++st) {
        const int c3 = c * 128 + kc * 32 + st * 16 + nl;
        const bool mok = Mk[c3] > 0.5f;
#pragma unroll
        for (int r = 0; r < 8; ++r) {
          const int diff = c3 - WWIN - (sub * 128 + w * 16 + hi * 8 + r);
          const bool ok = mok && (diff >= -WWIN) && (diff <= WWIN);
          s[st][r] = ok ? s[st][r] * scale : -1.0e9f;
        }
      }
      // ---- row max over 32 cols (half-wave butterfly; rows live per half)
      float mrow[8];
#pragma unroll
      for (int r = 0; r < 8; ++r) mrow[r] = fmaxf(s[0][r], s[1][r]);
#pragma unroll
      for (int off = 1; off < 16; off <<= 1)
#pragma unroll
        for (int r = 0; r < 8; ++r)
          mrow[r] = fmaxf(mrow[r], __shfl_xor(mrow[r], off, 32));
      // ---- online-softmax rescale of running state
#pragma unroll
      for (int r = 0; r < 8; ++r) {
        const float newm = fmaxf(rm[r], mrow[r]);
        const float alpha = __expf(rm[r] - newm);
        rm[r] = newm;
        rs[r] *= alpha;
#pragma unroll
        for (int j = 0; j < 4; ++j) o[j][r] *= alpha;
      }
      // ---- P = exp(S - m), row sums
      float psum[8];
#pragma unroll
      for (int r = 0; r < 8; ++r) {
        const float p0 = __expf(s[0][r] - rm[r]);
        const float p1 = __expf(s[1][r] - rm[r]);
        s[0][r] = p0; s[1][r] = p1;
        psum[r] = p0 + p1;
      }
#pragma unroll
      for (int off = 1; off < 16; off <<= 1)
#pragma unroll
        for (int r = 0; r < 8; ++r) psum[r] += __shfl_xor(psum[r], off, 32);
#pragma unroll
      for (int r = 0; r < 8; ++r) rs[r] += psum[r];
      // ---- re-layout P (C layout -> A fragment) through per-wave LDS scratch
      bf16_t* pw = &Ps[w * 640];
#pragma unroll
      for (int st = 0; st < 2; ++st)
#pragma unroll
        for (int r = 0; r < 8; ++r)
          pw[(hi * 8 + r) * 40 + st * 16 + nl] = (bf16_t)s[st][r];
      asm volatile("s_wait_dscnt 0x0" ::: "memory");  // same-wave DS RAW
      v16bf pf = ldsA_frag(pw, 40);
      // ---- O += P @ V  (one WMMA per 16-wide d tile)
#pragma unroll
      for (int j = 0; j < 4; ++j) {
        v16bf bv = ldsB_frag(&Vs[(j * 16) * 136 + kc * 32], 136);
        o[j] = wmma_bf16(pf, bv, o[j]);
      }
    }
  }

  // ---- finalize: divide by row sums, write ctx in [S, HID] layout
#pragma unroll
  for (int j = 0; j < 4; ++j)
#pragma unroll
    for (int r = 0; r < 8; ++r) {
      const int row = nch * WWIN + sub * 128 + w * 16 + hi * 8 + r;
      ctx[(size_t)row * HID + h * DHEAD + j * 16 + nl] = o[j][r] / rs[r];
    }
}

// ---------------------------------------------------------------------------
// Embedding gather + LayerNorm. One block per token row.
// ---------------------------------------------------------------------------
__global__ __launch_bounds__(256) void embed_ln_kernel(
    const int* __restrict__ ids, const float* __restrict__ wemb,
    const float* __restrict__ pemb, const float* __restrict__ lns,
    const float* __restrict__ lnb, float* __restrict__ x) {
  __shared__ float red[256];
  const int row = blockIdx.x;
  const int tid = threadIdx.x;
  const int id = ids[row];
  float v[3];
#pragma unroll
  for (int e = 0; e < 3; ++e) {
    const int cc = tid + e * 256;
    v[e] = wemb[(size_t)id * HID + cc] + pemb[(size_t)row * HID + cc];
  }
  const float mu = block_sum(v[0] + v[1] + v[2], red) * (1.f / HID);
  float vs = 0.f;
#pragma unroll
  for (int e = 0; e < 3; ++e) { const float d = v[e] - mu; vs += d * d; }
  const float inv = rsqrtf(block_sum(vs, red) * (1.f / HID) + 1e-5f);
#pragma unroll
  for (int e = 0; e < 3; ++e) {
    const int cc = tid + e * 256;
    x[(size_t)row * HID + cc] = (v[e] - mu) * inv * lns[cc] + lnb[cc];
  }
}

// ---------------------------------------------------------------------------
// Residual add + LayerNorm (safe in-place: row-independent).
// ---------------------------------------------------------------------------
__global__ __launch_bounds__(256) void add_ln_kernel(
    const float* __restrict__ xin, const float* __restrict__ yin,
    const float* __restrict__ lns, const float* __restrict__ lnb,
    float* __restrict__ xout) {
  __shared__ float red[256];
  const int row = blockIdx.x;
  const int tid = threadIdx.x;
  float v[3];
#pragma unroll
  for (int e = 0; e < 3; ++e) {
    const int cc = tid + e * 256;
    v[e] = xin[(size_t)row * HID + cc] + yin[(size_t)row * HID + cc];
  }
  const float mu = block_sum(v[0] + v[1] + v[2], red) * (1.f / HID);
  float vs = 0.f;
#pragma unroll
  for (int e = 0; e < 3; ++e) { const float d = v[e] - mu; vs += d * d; }
  const float inv = rsqrtf(block_sum(vs, red) * (1.f / HID) + 1e-5f);
#pragma unroll
  for (int e = 0; e < 3; ++e) {
    const int cc = tid + e * 256;
    xout[(size_t)row * HID + cc] = (v[e] - mu) * inv * lns[cc] + lnb[cc];
  }
}

// ---------------------------------------------------------------------------
// Classifier head: logits = pooled @ Wh + bh, plus focal-loss accumulation.
// ---------------------------------------------------------------------------
__global__ __launch_bounds__(256) void head_kernel(
    const float* __restrict__ x, const float* __restrict__ Wh,
    const float* __restrict__ bh, const float* __restrict__ labels,
    float* logits, float* loss, int N, float inv_n) {
  __shared__ float pooled[HID];
  __shared__ float red[256];
  const int tid = threadIdx.x;
  for (int cc = tid; cc < HID; cc += 256) pooled[cc] = x[cc];
  __syncthreads();
  const int n = blockIdx.x * 256 + tid;
  float contrib = 0.f;
  if (n < N) {
    float a = bh[n];
    for (int k2 = 0; k2 < HID; ++k2) a = fmaf(pooled[k2], Wh[(size_t)k2 * N + n], a);
    if (logits) logits[n] = a;
    const float t = labels[n];
    const float bce = fmaxf(a, 0.f) - a * t + log1pf(__expf(-fabsf(a)));
    const float pt = __expf(-bce);
    const float om = 1.f - pt;
    contrib = om * om * bce;
  }
  red[tid] = contrib;
  __syncthreads();
  for (int off = 128; off > 0; off >>= 1) {
    if (tid < off) red[tid] += red[tid + off];
    __syncthreads();
  }
  if (tid == 0) atomicAdd(loss, red[0] * inv_n);
}

__global__ void zero_kernel(float* p) {
  if (threadIdx.x == 0 && blockIdx.x == 0) p[0] = 0.f;
}

// ---------------------------------------------------------------------------
// Launch sequence
// ---------------------------------------------------------------------------
extern "C" void kernel_launch(void* const* d_in, const int* in_sizes, int n_in,
                              void* d_out, int out_size, void* d_ws, size_t ws_size,
                              hipStream_t stream) {
  const int* ids = (const int*)d_in[0];
  const float* amask = (const float*)d_in[1];
  const float* icd_lab = (const float*)d_in[2];
  const float* cpt_lab = (const float*)d_in[3];
  const float* ch_lab = (const float*)d_in[4];
  const float* wemb = (const float*)d_in[5];
  const float* pemb = (const float*)d_in[6];
  const float* elns = (const float*)d_in[7];
  const float* elnb = (const float*)d_in[8];
  const float* Wq = (const float*)d_in[9];
  const float* bq = (const float*)d_in[10];
  const float* Wk = (const float*)d_in[11];
  const float* bk = (const float*)d_in[12];
  const float* Wv = (const float*)d_in[13];
  const float* bv = (const float*)d_in[14];
  const float* Wo = (const float*)d_in[15];
  const float* bo = (const float*)d_in[16];
  const float* alns = (const float*)d_in[17];
  const float* alnb = (const float*)d_in[18];
  const float* Wi = (const float*)d_in[19];
  const float* bi = (const float*)d_in[20];
  const float* Wo2 = (const float*)d_in[21];
  const float* bo2 = (const float*)d_in[22];
  const float* flns = (const float*)d_in[23];
  const float* flnb = (const float*)d_in[24];
  const float* Wicd = (const float*)d_in[25];
  const float* bicd = (const float*)d_in[26];
  const float* Wcpt = (const float*)d_in[27];
  const float* bcpt = (const float*)d_in[28];
  const float* Wch = (const float*)d_in[29];
  const float* bch = (const float*)d_in[30];

  float* ws = (float*)d_ws;
  const size_t SH = (size_t)S_LEN * HID;
  float* x = ws;
  float* q = ws + SH;
  float* k = ws + 2 * SH;
  float* v = ws + 3 * SH;
  float* ctx = ws + 4 * SH;
  float* tmp = ws + 5 * SH;
  float* hbuf = ws + 6 * SH;  // S_LEN * FF = 4*SH floats

  float* out = (float*)d_out;

  embed_ln_kernel<<<S_LEN, 256, 0, stream>>>(ids, wemb, pemb, elns, elnb, x);

  const dim3 gProj(HID / 128, S_LEN / 128);
  const dim3 gFF(FF / 128, S_LEN / 128);
  for (int l = 0; l < NLAYER; ++l) {
    const size_t wofs = (size_t)l * HID * HID;
    gemm_kernel<<<gProj, 256, 0, stream>>>(x, Wq + wofs, bq + l * HID, q,
                                           S_LEN, HID, HID, 0);
    gemm_kernel<<<gProj, 256, 0, stream>>>(x, Wk + wofs, bk + l * HID, k,
                                           S_LEN, HID, HID, 0);
    gemm_kernel<<<gProj, 256, 0, stream>>>(x, Wv + wofs, bv + l * HID, v,
                                           S_LEN, HID, HID, 0);
    band_attn_kernel<<<dim3(2, NCHUNK, NHEAD), 256, 0, stream>>>(q, k, v, amask, ctx);
    gemm_kernel<<<gProj, 256, 0, stream>>>(ctx, Wo + wofs, bo + l * HID, tmp,
                                           S_LEN, HID, HID, 0);
    add_ln_kernel<<<S_LEN, 256, 0, stream>>>(x, tmp, alns + l * HID, alnb + l * HID, x);
    gemm_kernel<<<gFF, 256, 0, stream>>>(x, Wi + (size_t)l * HID * FF, bi + l * FF,
                                         hbuf, S_LEN, FF, HID, 1);
    gemm_kernel<<<gProj, 256, 0, stream>>>(hbuf, Wo2 + (size_t)l * FF * HID,
                                           bo2 + l * HID, tmp, S_LEN, HID, FF, 0);
    add_ln_kernel<<<S_LEN, 256, 0, stream>>>(x, tmp, flns + l * HID, flnb + l * HID, x);
  }

  zero_kernel<<<1, 32, 0, stream>>>(out);
  head_kernel<<<(NUM_ICD + 255) / 256, 256, 0, stream>>>(
      x, Wicd, bicd, icd_lab, out + 1, out, NUM_ICD, 1.f / NUM_ICD);
  head_kernel<<<(NUM_CPT + 255) / 256, 256, 0, stream>>>(
      x, Wcpt, bcpt, cpt_lab, out + 1 + NUM_ICD, out, NUM_CPT, 1.f / NUM_CPT);
  head_kernel<<<1, 256, 0, stream>>>(
      x, Wch, bch, ch_lab, nullptr, out, NUM_CH, 1.f / NUM_CH);

  (void)in_sizes; (void)n_in; (void)out_size; (void)ws_size;
}
